// CrossLevelAttention_28621662061172
// MI455X (gfx1250) — matrix operand order repaired and checked by reading, not compiled, not run on hardware
//
#include <hip/hip_runtime.h>
#include <hip/hip_bf16.h>

typedef __attribute__((ext_vector_type(16))) _Float16 v16h;
typedef __attribute__((ext_vector_type(8)))  float    v8f;
typedef _Float16 half_t;

// ---------------------------------------------------------------------------
// WMMA operand loaders (wave32, V_WMMA_F32_16X16X32_F16; A/B use this file's contiguous k order)
// A (16x32, row-major source): lane holds row m=row0+(lane&15);
//   K = k0 + (lane&16?16:0) + e (contiguous, matching B).
// B (32x16): lane holds column n of B; K elems contiguous:
//   K = k0 + (lane&16?16:0) + e.
// C/D (16x16 f32): vgpr r, lane L -> m = r + 8*(L>=16), n = L&15.
// ---------------------------------------------------------------------------
__device__ __forceinline__ v16h load_opA(const half_t* base, int ld, int row0,
                                         int col0, int lane) {
  int m  = row0 + (lane & 15);
  int k0 = col0 + ((lane & 16) ? 16 : 0);
  const half_t* p = base + (size_t)m * ld + k0;
  v16h a;
#pragma unroll
  for (int e = 0; e < 8; ++e) a[e] = p[e];
#pragma unroll
  for (int e = 0; e < 8; ++e) a[e + 8] = p[e + 8];
  return a;
}

// B operand from a row-major [N,K] weight matrix (computes X * W^T)
__device__ __forceinline__ v16h load_opB_rows(const half_t* base, int ld,
                                              int k0, int n0, int lane) {
  int n  = n0 + (lane & 15);
  int kb = k0 + ((lane & 16) ? 16 : 0);
  const half_t* p = base + (size_t)n * ld + kb;
  v16h b;
#pragma unroll
  for (int e = 0; e < 16; ++e) b[e] = p[e];
  return b;
}

// B operand from a row-major [K,N] matrix (true columns, strided)
__device__ __forceinline__ v16h load_opB_cols(const half_t* base, int ld,
                                              int k0, int n0, int lane) {
  int n  = n0 + (lane & 15);
  int kb = k0 + ((lane & 16) ? 16 : 0);
  const half_t* p = base + (size_t)kb * ld + n;
  v16h b;
#pragma unroll
  for (int e = 0; e < 16; ++e) b[e] = p[(size_t)e * ld];
  return b;
}

#define WMMA_F16(a, b, c) \
  __builtin_amdgcn_wmma_f32_16x16x32_f16(false, (a), false, (b), (short)0, (c), false, false)

// ---------------------------------------------------------------------------
// Problem constants
// ---------------------------------------------------------------------------
#define BATCH   16
#define H       110
#define WID     110
#define NTOK    (BATCH * H * WID)       // 193600
#define NTOK_KV (BATCH * 55 * 55)       // 48400
#define NWIN    (BATCH * 16 * 16)       // 4096 windows of 7x7

// ---------------------------------------------------------------------------
// Kernel 1: fuse deconv weights into K/V 1x1 weights, convert to f16,
//           fuse biases, zero BN accumulators.
//   KW[kl][o][c] = sum_o2 k_w[o,o2] * deconv_w[c,o2,dk,dl]
// ---------------------------------------------------------------------------
__global__ void __launch_bounds__(256) prep_kernel(
    const float* __restrict__ dw, const float* __restrict__ db,
    const float* __restrict__ kw, const float* __restrict__ kb,
    const float* __restrict__ vw, const float* __restrict__ vb,
    const float* __restrict__ qw, const float* __restrict__ ow,
    half_t* __restrict__ KWh, half_t* __restrict__ VWh,
    half_t* __restrict__ qwh, half_t* __restrict__ owh,
    float* __restrict__ Kb, float* __restrict__ Vb, float* __restrict__ sums) {
  int idx = blockIdx.x * 256 + threadIdx.x;
  if (idx < 98304) {                                   // KWh: 4*128*192
    int kl = idx / (128 * 192), rem = idx % (128 * 192);
    int o = rem / 192, c = rem % 192;
    int dk = kl >> 1, dl = kl & 1;
    float s = 0.f;
    for (int o2 = 0; o2 < 192; ++o2)
      s += kw[o * 192 + o2] * dw[((c * 192 + o2) * 2 + dk) * 2 + dl];
    KWh[idx] = (half_t)s;
  } else if (idx < 98304 + 147456) {                   // VWh: 4*192*192
    int j = idx - 98304;
    int kl = j / (192 * 192), rem = j % (192 * 192);
    int o = rem / 192, c = rem % 192;
    int dk = kl >> 1, dl = kl & 1;
    float s = 0.f;
    for (int o2 = 0; o2 < 192; ++o2)
      s += vw[o * 192 + o2] * dw[((c * 192 + o2) * 2 + dk) * 2 + dl];
    VWh[j] = (half_t)s;
  } else if (idx < 245760 + 12288) {                   // q_w -> f16
    int j = idx - 245760;
    qwh[j] = (half_t)qw[j];
  } else if (idx < 258048 + 18432) {                   // out_w -> f16
    int j = idx - 258048;
    owh[j] = (half_t)ow[j];
  } else if (idx < 276480 + 128) {                     // fused K bias
    int o = idx - 276480;
    float s = kb[o];
    for (int o2 = 0; o2 < 192; ++o2) s += kw[o * 192 + o2] * db[o2];
    Kb[o] = s;
  } else if (idx < 276608 + 192) {                     // fused V bias
    int o = idx - 276608;
    float s = vb[o];
    for (int o2 = 0; o2 < 192; ++o2) s += vw[o * 192 + o2] * db[o2];
    Vb[o] = s;
  } else if (idx < 276800 + 192) {                     // zero sum/sumsq
    sums[idx - 276800] = 0.f;
  }
}
#define PREP_THREADS 276992

// ---------------------------------------------------------------------------
// Kernel 2: Q = low_feat * q_w^T + q_b   (WMMA, f16 out, token-major [tok,128])
// ---------------------------------------------------------------------------
__global__ void __launch_bounds__(256) q_proj_kernel(
    const float* __restrict__ low, const half_t* __restrict__ qwh,
    const float* __restrict__ qb, half_t* __restrict__ Qg) {
  __shared__ half_t As[64 * 96];
  int tid = threadIdx.x;
  for (int i = tid; i < 64 * 96; i += 256) {
    int c = i >> 6, t = i & 63;
    int token = blockIdx.x * 64 + t;          // exact: 3025*64 == NTOK
    int b = token / 12100, p = token % 12100;
    As[t * 96 + c] = (half_t)low[((size_t)b * 96 + c) * 12100 + p];
  }
  __syncthreads();
  int wave = tid >> 5, lane = tid & 31;
  for (int job = wave; job < 32; job += 8) {  // 4 Mtiles x 8 Ntiles
    int tm = job >> 3, tn = job & 7;
    v8f acc = {0, 0, 0, 0, 0, 0, 0, 0};
    for (int kk = 0; kk < 96; kk += 32) {
      v16h a = load_opA(As, 96, tm * 16, kk, lane);
      v16h b = load_opB_rows(qwh, 96, kk, tn * 16, lane);
      acc = WMMA_F16(a, b, acc);
    }
    int n = tn * 16 + (lane & 15);
    float bias = qb[n];
#pragma unroll
    for (int r = 0; r < 8; ++r) {
      int m = tm * 16 + r + ((lane & 16) ? 8 : 0);
      int token = blockIdx.x * 64 + m;
      Qg[(size_t)token * 128 + n] = (half_t)(acc[r] + bias);
    }
  }
}

// ---------------------------------------------------------------------------
// Kernel 3: K/V via fused deconv+1x1 weights. grid.y = (dk,dl) parity.
//   output pixel (2h+dk, 2w+dl) <- high_feat[:, h, w] * KW[kl]^T
// ---------------------------------------------------------------------------
__global__ void __launch_bounds__(256) kv_proj_kernel(
    const float* __restrict__ high, const half_t* __restrict__ KWh,
    const half_t* __restrict__ VWh, const float* __restrict__ Kb,
    const float* __restrict__ Vb, half_t* __restrict__ Kg,
    half_t* __restrict__ Vg) {
  __shared__ half_t As[64 * 192];
  int tid = threadIdx.x;
  int kl = blockIdx.y, dk = kl >> 1, dl = kl & 1;
  for (int i = tid; i < 64 * 192; i += 256) {
    int c = i >> 6, t = i & 63;
    int token = blockIdx.x * 64 + t;
    float v = 0.f;
    if (token < NTOK_KV) {
      int b = token / 3025, r2 = token % 3025;
      v = high[((size_t)b * 192 + c) * 3025 + r2];
    }
    As[t * 192 + c] = (half_t)v;
  }
  __syncthreads();
  int wave = tid >> 5, lane = tid & 31;
  const half_t* Wk = KWh + (size_t)kl * 128 * 192;
  const half_t* Wv = VWh + (size_t)kl * 192 * 192;
  for (int job = wave; job < 80; job += 8) {
    bool isK = job < 32;
    int tm, tn;
    const half_t* Wp;
    if (isK) { tm = job >> 3; tn = job & 7; Wp = Wk; }
    else     { int j = job - 32; tm = j / 12; tn = j % 12; Wp = Wv; }
    v8f acc = {0, 0, 0, 0, 0, 0, 0, 0};
    for (int kk = 0; kk < 192; kk += 32) {
      v16h a = load_opA(As, 192, tm * 16, kk, lane);
      v16h b = load_opB_rows(Wp, 192, kk, tn * 16, lane);
      acc = WMMA_F16(a, b, acc);
    }
    int n = tn * 16 + (lane & 15);
    float bias = isK ? Kb[n] : Vb[n];
#pragma unroll
    for (int r = 0; r < 8; ++r) {
      int m = tm * 16 + r + ((lane & 16) ? 8 : 0);
      int token = blockIdx.x * 64 + m;
      if (token < NTOK_KV) {
        int b = token / 3025, r2 = token % 3025;
        int h = r2 / 55, w = r2 % 55;
        size_t dst = ((size_t)b * 110 + (2 * h + dk)) * 110 + (2 * w + dl);
        if (isK) Kg[dst * 128 + n] = (half_t)(acc[r] + bias);
        else     Vg[dst * 192 + n] = (half_t)(acc[r] + bias);
      }
    }
  }
}

// ---------------------------------------------------------------------------
// Kernel 4: windowed attention + out-projection. One wave per 7x7 window.
//   49 tokens padded to 64; reflect-pad gather via GLOBAL_LOAD_ASYNC_TO_LDS;
//   S=QK^T, softmax, O=PV, P2 = O*out_w^T + out_b; pre-BN out + channel stats.
// ---------------------------------------------------------------------------
__global__ void __launch_bounds__(32) attn_kernel(
    const half_t* __restrict__ Qg, const half_t* __restrict__ Kg,
    const half_t* __restrict__ Vg, const half_t* __restrict__ owh,
    const float* __restrict__ ob, const float* __restrict__ scale_p,
    float* __restrict__ preBN, float* __restrict__ sums) {
  extern __shared__ char smem[];
  float*  Sf  = (float*)smem;                    // 64*64 f32
  float*  red = Sf + 64 * 64;                    // 192 f32
  half_t* Qs  = (half_t*)(red + 192);            // 64*128
  half_t* Ks  = Qs + 64 * 128;                   // 64*128
  half_t* Vs  = Ks + 64 * 128;                   // 64*192
  half_t* Sh  = Vs + 64 * 192;                   // 64*64
  half_t* Os  = Sh + 64 * 64;                    // 64*192

  int lane = threadIdx.x;
  int b = blockIdx.x >> 8, win = blockIdx.x & 255;
  int wy = (win >> 4) * 7, wx = (win & 15) * 7;

  for (int i = lane; i < 192; i += 32) red[i] = 0.f;

  // --- Async gather of Q/K/V rows through reflect padding (i>=110 -> 218-i).
  // CDNA5 GLOBAL_LOAD_ASYNC_TO_LDS_B128: per-lane 16B global->LDS DMA,
  // tracked by ASYNCcnt (no VGPR staging). Q/K rows: 16 lanes x 16B each;
  // V row: 24 lanes x 16B. LDS byte offsets are relative to the dynamic-LDS
  // base (no static LDS in this kernel).
  for (int t = 0; t < 49; ++t) {
    int hp = wy + t / 7, wp = wx + t % 7;
    int h = hp < 110 ? hp : 218 - hp;
    int w = wp < 110 ? wp : 218 - wp;
    size_t tok = ((size_t)b * 110 + h) * 110 + w;
    {
      int l16 = lane & 15;
      const half_t* g = (lane < 16) ? (Qg + tok * 128 + l16 * 8)
                                    : (Kg + tok * 128 + l16 * 8);
      const half_t* d = (lane < 16) ? (Qs + t * 128 + l16 * 8)
                                    : (Ks + t * 128 + l16 * 8);
      unsigned loff = (unsigned)((const char*)d - smem);
      asm volatile("global_load_async_to_lds_b128 %0, %1, off"
                   :: "v"(loff), "v"((unsigned long long)(uintptr_t)g)
                   : "memory");
    }
    if (lane < 24) {
      const half_t* g = Vg + tok * 192 + lane * 8;
      unsigned loff = (unsigned)((const char*)(Vs + t * 192 + lane * 8) - smem);
      asm volatile("global_load_async_to_lds_b128 %0, %1, off"
                   :: "v"(loff), "v"((unsigned long long)(uintptr_t)g)
                   : "memory");
    }
  }
  // Zero the 15 pad rows (normal DS stores; disjoint from async region)
  for (int t = 49; t < 64; ++t) {
    uint2 z; z.x = 0; z.y = 0;
    ((uint2*)(Qs + t * 128))[lane] = z;
    ((uint2*)(Ks + t * 128))[lane] = z;
    unsigned* vd = (unsigned*)(Vs + t * 192);
    vd[lane] = 0; vd[lane + 32] = 0; vd[lane + 64] = 0;
  }
  asm volatile("s_wait_asynccnt 0x0" ::: "memory");
  __syncthreads();

  // S = Q * K^T * scale  (16 tiles, K-loop = 128/32)
  float scl = *scale_p;
  for (int tm = 0; tm < 4; ++tm)
    for (int tn = 0; tn < 4; ++tn) {
      v8f acc = {0, 0, 0, 0, 0, 0, 0, 0};
      for (int kk = 0; kk < 128; kk += 32) {
        v16h a = load_opA(Qs, 128, tm * 16, kk, lane);
        v16h bm = load_opB_rows(Ks, 128, kk, tn * 16, lane);
        acc = WMMA_F16(a, bm, acc);
      }
      int n = tn * 16 + (lane & 15);
#pragma unroll
      for (int r = 0; r < 8; ++r) {
        int m = tm * 16 + r + ((lane & 16) ? 8 : 0);
        Sf[m * 64 + n] = acc[r] * scl;
      }
    }
  __syncthreads();

  // Masked softmax over key tokens 0..48; each lane handles 2 rows
  for (int rr = 0; rr < 2; ++rr) {
    int r = lane + rr * 32;
    float mx = -1e30f;
    for (int c = 0; c < 49; ++c) mx = fmaxf(mx, Sf[r * 64 + c]);
    float s = 0.f;
    for (int c = 0; c < 49; ++c) s += __expf(Sf[r * 64 + c] - mx);
    float inv = 1.0f / s;
    for (int c = 0; c < 49; ++c) Sh[r * 64 + c] = (half_t)(__expf(Sf[r * 64 + c] - mx) * inv);
    for (int c = 49; c < 64; ++c) Sh[r * 64 + c] = (half_t)0.f;
  }
  __syncthreads();

  // O = softmax(S) * V   (4x12 tiles, K-loop = 64/32)
  for (int tm = 0; tm < 4; ++tm)
    for (int tn = 0; tn < 12; ++tn) {
      v8f acc = {0, 0, 0, 0, 0, 0, 0, 0};
      for (int kk = 0; kk < 64; kk += 32) {
        v16h a = load_opA(Sh, 64, tm * 16, kk, lane);
        v16h bm = load_opB_cols(Vs, 192, kk, tn * 16, lane);
        acc = WMMA_F16(a, bm, acc);
      }
      int n = tn * 16 + (lane & 15);
#pragma unroll
      for (int r = 0; r < 8; ++r) {
        int m = tm * 16 + r + ((lane & 16) ? 8 : 0);
        Os[m * 192 + n] = (half_t)acc[r];
      }
    }
  __syncthreads();

  // P = O * out_w^T + out_b; write valid tokens + BN partial stats
  for (int tm = 0; tm < 4; ++tm)
    for (int tn = 0; tn < 6; ++tn) {
      v8f acc = {0, 0, 0, 0, 0, 0, 0, 0};
      for (int kk = 0; kk < 192; kk += 32) {
        v16h a = load_opA(Os, 192, tm * 16, kk, lane);
        v16h bm = load_opB_rows(owh, 192, kk, tn * 16, lane);
        acc = WMMA_F16(a, bm, acc);
      }
      int n = tn * 16 + (lane & 15);
      float bias = ob[n];
      float ls = 0.f, ls2 = 0.f;
#pragma unroll
      for (int r = 0; r < 8; ++r) {
        int m = tm * 16 + r + ((lane & 16) ? 8 : 0);
        if (m < 49) {
          int hp = wy + m / 7, wp = wx + m % 7;
          if (hp < 110 && wp < 110) {
            float v = acc[r] + bias;
            preBN[(((size_t)b * 110 + hp) * 110 + wp) * 96 + n] = v;
            ls += v; ls2 += v * v;
          }
        }
      }
      atomicAdd(&red[n], ls);
      atomicAdd(&red[96 + n], ls2);
    }
  __syncthreads();
  for (int i = lane; i < 192; i += 32) atomicAdd(&sums[i], red[i]);
}
#define ATTN_SMEM ((64 * 64 + 192) * 4 + (64 * 128 * 2 + 64 * 192 * 2 + 64 * 64) * 2)

// ---------------------------------------------------------------------------
// Kernel 5: BatchNorm (batch stats) + affine + ReLU, write NCHW fp32 output.
// ---------------------------------------------------------------------------
__global__ void __launch_bounds__(256) bn_kernel(
    const float* __restrict__ preBN, const float* __restrict__ sums,
    const float* __restrict__ gamma, const float* __restrict__ beta,
    float* __restrict__ out) {
  size_t idx = (size_t)blockIdx.x * 256 + threadIdx.x;
  const size_t total = (size_t)BATCH * 96 * 12100;
  if (idx >= total) return;
  int p = (int)(idx % 12100);
  int c = (int)((idx / 12100) % 96);
  int b = (int)(idx / (12100ull * 96));
  float x = preBN[((size_t)b * 12100 + p) * 96 + c];
  const float invN = 1.0f / (float)NTOK;
  float mean = sums[c] * invN;
  float var  = sums[96 + c] * invN - mean * mean;
  float y = (x - mean) * rsqrtf(var + 1e-5f) * gamma[c] + beta[c];
  out[idx] = fmaxf(y, 0.f);
}

// ---------------------------------------------------------------------------
extern "C" void kernel_launch(void* const* d_in, const int* in_sizes, int n_in,
                              void* d_out, int out_size, void* d_ws, size_t ws_size,
                              hipStream_t stream) {
  const float* low   = (const float*)d_in[0];
  const float* high  = (const float*)d_in[1];
  const float* dw    = (const float*)d_in[2];
  const float* db    = (const float*)d_in[3];
  const float* qw    = (const float*)d_in[4];
  const float* qb    = (const float*)d_in[5];
  const float* kw    = (const float*)d_in[6];
  const float* kb    = (const float*)d_in[7];
  const float* vw    = (const float*)d_in[8];
  const float* vb    = (const float*)d_in[9];
  const float* scale = (const float*)d_in[10];
  const float* ow    = (const float*)d_in[11];
  const float* ob    = (const float*)d_in[12];
  const float* gamma = (const float*)d_in[13];
  const float* beta  = (const float*)d_in[14];

  char* ws = (char*)d_ws;
  size_t off = 0;
  auto take = [&](size_t bytes) {
    char* p = ws + off;
    off = (off + bytes + 255) & ~(size_t)255;
    return p;
  };
  half_t* KWh   = (half_t*)take(4 * 128 * 192 * sizeof(half_t));
  half_t* VWh   = (half_t*)take(4 * 192 * 192 * sizeof(half_t));
  half_t* qwh   = (half_t*)take(128 * 96 * sizeof(half_t));
  half_t* owh   = (half_t*)take(96 * 192 * sizeof(half_t));
  float*  Kbf   = (float*)take(128 * sizeof(float));
  float*  Vbf   = (float*)take(192 * sizeof(float));
  float*  sums  = (float*)take(192 * sizeof(float));          // sum[96]+sumsq[96]
  half_t* Qg    = (half_t*)take((size_t)NTOK * 128 * sizeof(half_t));
  half_t* Kg    = (half_t*)take((size_t)NTOK * 128 * sizeof(half_t));
  half_t* Vg    = (half_t*)take((size_t)NTOK * 192 * sizeof(half_t));
  float*  preBN = (float*)take((size_t)NTOK * 96 * sizeof(float));
  (void)ws_size; (void)in_sizes; (void)n_in; (void)out_size;

  prep_kernel<<<(PREP_THREADS + 255) / 256, 256, 0, stream>>>(
      dw, db, kw, kb, vw, vb, qw, ow, KWh, VWh, qwh, owh, Kbf, Vbf, sums);

  q_proj_kernel<<<NTOK / 64, 256, 0, stream>>>(low, qwh, qb, Qg);

  kv_proj_kernel<<<dim3((NTOK_KV + 63) / 64, 4), 256, 0, stream>>>(
      high, KWh, VWh, Kbf, Vbf, Kg, Vg);

  attn_kernel<<<NWIN, 32, ATTN_SMEM, stream>>>(
      Qg, Kg, Vg, owh, ob, scale, preBN, sums);

  const size_t total = (size_t)BATCH * 96 * 12100;
  bn_kernel<<<(unsigned)((total + 255) / 256), 256, 0, stream>>>(
      preBN, sums, gamma, beta, (float*)d_out);
}
